// MultiLayerLSTM_65618510348831
// MI455X (gfx1250) — matrix operand order repaired and checked by reading, not compiled, not run on hardware
//
#include <hip/hip_runtime.h>
#include <hip/hip_bf16.h>
#include <stdint.h>

typedef __attribute__((ext_vector_type(16))) _Float16 v16h;
typedef __attribute__((ext_vector_type(8)))  float    v8f;

// global (device memory) address-space typed views, so loads derived from the
// asm-opaqued pointer still lower to global_load_* (LOADcnt) instead of
// flat_load_* (LOADcnt + DScnt, competes with the LDS A-fragment stream).
typedef __attribute__((address_space(1))) const _Float16 gchalf;
typedef __attribute__((address_space(1))) const v16h     gcv16h;

#define B_  128
#define T_  512
#define I_  64
#define H_  512
#define LDP 520          // padded LDS row stride (halves): 1040B -> stride-4 banks, conflict-free
#define XTP 72           // padded x-tile stride (halves)

// ---------------------------------------------------------------------------
// Weight pre-pack: f32 [K,N] row-major -> f16 per-lane WMMA B-fragments.
// B-matrix 32x16 (KxN), wave32 layout: lane = (n&15) | (k&8 ? 16:0),
// within lane halves ordered so V0..3 hold K = 0..7, V4..7 K = 16..23 (+8 for lanes 16-31).
// packed[((kt*NT + nt)*32 + lane)*16 + e],  e = (kk&7) | ((kk>>4)<<3), lane = nn | (((kk>>3)&1)<<4)
// ---------------------------------------------------------------------------
__global__ void pack_weights_kernel(const float* __restrict__ W,
                                    _Float16* __restrict__ out, int K, int N) {
    int idx = blockIdx.x * blockDim.x + threadIdx.x;
    if (idx >= K * N) return;
    int k = idx / N, n = idx - k * N;
    int kt = k >> 5, kk = k & 31;
    int nt = n >> 4, nn = n & 15;
    int g    = (kk >> 3) & 1;
    int e    = (kk & 7) | ((kk >> 4) << 3);
    int lane = nn | (g << 4);
    int NT   = N >> 4;
    out[((((size_t)kt * NT + nt) * 32 + lane) << 4) + e] = (_Float16)W[idx];
}

// ---------------------------------------------------------------------------
// A-fragment loader: 16x32 (MxK) f16 tile from LDS, per ISA 7.12.2:
// lanes 0-15: M=lane, V0..3 K=0..7, V4..7 K=16..23 ; lanes 16-31: K +8
// K pairs (2v,2v+1) are contiguous -> dword reads, merged into 2x ds_load_b128.
// ---------------------------------------------------------------------------
__device__ __forceinline__ v16h load_a_frag(const _Float16* __restrict__ buf,
                                            int stride, int kbase, int lane) {
    const int M = lane & 15;
    const int g = lane >> 4;
    union { uint32_t u[8]; v16h v; } a;
    const _Float16* row = buf + M * stride + kbase + (g ? 8 : 0);
#pragma unroll
    for (int r = 0; r < 8; ++r) {
        int K = 2 * (r & 3) + ((r & 4) ? 16 : 0);
        a.u[r] = *(const uint32_t*)(row + K);
    }
    return a.v;
}

// Accumulate  z[16 x (4*512)]  tiles owned by this wave:
// tile index q = gt*2 + hb  ->  columns  gt*512 + w*32 + hb*16 .. +15
// The asm barrier makes the (global-typed) weight base opaque per call so the
// 512-step time loop cannot hoist (and then spill) the weight fragments: they
// stay as in-loop global_load_b128 hits on the L2-resident packed image.
__device__ __forceinline__ void mm_acc(v8f acc[8],
                                       const _Float16* __restrict__ abuf, int astride,
                                       int nkt,
                                       const _Float16* __restrict__ wpack,
                                       int w, int lane) {
    gchalf* wg = (gchalf*)wpack;                 // addrspacecast generic -> global
    asm volatile("" : "+s"(wg));                 // defeat LICM across timesteps
    gchalf* base = wg + ((size_t)w * 1024 + (size_t)lane * 16);
    // fragment offset from base (halves): kt*65536 + gt*16384 + hb*512
    for (int kt = 0; kt < nkt; ++kt) {
        v16h a = load_a_frag(abuf, astride, kt * 32, lane);
        gchalf* kb = base + (size_t)kt * 65536;
        v16h bf[8];
#pragma unroll
        for (int gt = 0; gt < 4; ++gt)
#pragma unroll
            for (int hb = 0; hb < 2; ++hb)
                bf[gt * 2 + hb] = *(gcv16h*)(kb + gt * 16384 + hb * 512);
#pragma unroll
        for (int q = 0; q < 8; ++q)
            acc[q] = __builtin_amdgcn_wmma_f32_16x16x32_f16(
                false, a, false, bf[q], (short)0, acc[q], false, false);
    }
}

__device__ __forceinline__ float sigf(float x) { return 1.f / (1.f + __expf(-x)); }

// Gates + cell update + LayerNorm across H (cross-wave via LDS).
// acc lane layout (C 16x16 f32): VGPR r -> row r + 8*(lane>>4), col lane&15.
__device__ __forceinline__ void gate_ln(const v8f acc[8], float* crow,
                                        const float bi[2], const float bf[2],
                                        const float bg[2], const float bo[2],
                                        const float gam[2], const float bet[2],
                                        float* red_s, float* red_q,
                                        float* smean, float* srstd,
                                        int w, int lane, float hout[16]) {
    const int gg = lane >> 4;
#pragma unroll
    for (int hb = 0; hb < 2; ++hb) {
#pragma unroll
        for (int r = 0; r < 8; ++r) {
            float zi = acc[0 * 2 + hb][r] + bi[hb];
            float zf = acc[1 * 2 + hb][r] + bf[hb];
            float zg = acc[2 * 2 + hb][r] + bg[hb];
            float zo = acc[3 * 2 + hb][r] + bo[hb];
            float c  = sigf(zf) * crow[hb * 8 + r] + sigf(zi) * tanhf(zg);
            crow[hb * 8 + r] = c;
            hout[hb * 8 + r] = sigf(zo) * tanhf(c);
        }
    }
    // per-row partial sums over this wave's 32 hidden columns
    float rs[8], rq[8];
#pragma unroll
    for (int r = 0; r < 8; ++r) {
        float a0 = hout[r], a1 = hout[8 + r];
        float s = a0 + a1, q = a0 * a0 + a1 * a1;
#pragma unroll
        for (int m = 1; m < 16; m <<= 1) {
            s += __shfl_xor(s, m, 32);
            q += __shfl_xor(q, m, 32);
        }
        rs[r] = s; rq[r] = q;
    }
    if ((lane & 15) == 0) {
#pragma unroll
        for (int r = 0; r < 8; ++r) {
            int row = gg * 8 + r;
            red_s[w * 16 + row] = rs[r];
            red_q[w * 16 + row] = rq[r];
        }
    }
    __syncthreads();
    if (w == 0 && lane < 16) {                 // wave 0 finalizes mean/rstd
        float S = 0.f, Q = 0.f;
#pragma unroll
        for (int ww = 0; ww < 16; ++ww) {
            S += red_s[ww * 16 + lane];
            Q += red_q[ww * 16 + lane];
        }
        float m = S * (1.f / (float)H_);
        float v = Q * (1.f / (float)H_) - m * m;
        smean[lane] = m;
        srstd[lane] = rsqrtf(v + 1e-5f);
    }
    __syncthreads();
#pragma unroll
    for (int hb = 0; hb < 2; ++hb)
#pragma unroll
        for (int r = 0; r < 8; ++r) {
            int row = gg * 8 + r;
            hout[hb * 8 + r] =
                (hout[hb * 8 + r] - smean[row]) * srstd[row] * gam[hb] + bet[hb];
        }
}

// ---------------------------------------------------------------------------
// Persistent recurrence kernel: 8 blocks x 512 threads (16 waves).
// Block owns 16 batch rows end-to-end; c-state in registers; h-state in LDS f16.
// ---------------------------------------------------------------------------
__global__ __launch_bounds__(512, 1)
void lstm_kernel(const float* __restrict__ x,
                 const _Float16* __restrict__ wx0p, const _Float16* __restrict__ wh0p,
                 const _Float16* __restrict__ wx1p, const _Float16* __restrict__ wh1p,
                 const float* __restrict__ b0v, const float* __restrict__ g0v,
                 const float* __restrict__ be0v,
                 const float* __restrict__ b1v, const float* __restrict__ g1v,
                 const float* __restrict__ be1v,
                 float* __restrict__ hs) {
    __shared__ _Float16 sh0[16 * LDP];
    __shared__ _Float16 sh1[16 * LDP];
    __shared__ _Float16 sxt[16 * XTP];
    __shared__ float red_s0[16 * 16], red_q0[16 * 16];
    __shared__ float red_s1[16 * 16], red_q1[16 * 16];
    __shared__ float smean0[16], srstd0[16], smean1[16], srstd1[16];

    const int tid  = threadIdx.x;
    const int lane = tid & 31;
    const int w    = tid >> 5;
    const int gg   = lane >> 4;
    const int nn   = lane & 15;
    const int brow0 = blockIdx.x * 16;
    const int col0  = w * 32 + nn;

    for (int i = tid; i < 16 * LDP; i += 512) {
        sh0[i] = (_Float16)0.f;
        sh1[i] = (_Float16)0.f;
    }

    // per-lane constant parameters (bias / gamma / beta at owned columns)
    float bi0[2], bf0[2], bg0[2], bo0[2], bi1[2], bf1[2], bg1[2], bo1[2];
    float g0c[2], be0c[2], g1c[2], be1c[2];
#pragma unroll
    for (int hb = 0; hb < 2; ++hb) {
        int c = col0 + hb * 16;
        bi0[hb] = b0v[0 * H_ + c]; bf0[hb] = b0v[1 * H_ + c];
        bg0[hb] = b0v[2 * H_ + c]; bo0[hb] = b0v[3 * H_ + c];
        bi1[hb] = b1v[0 * H_ + c]; bf1[hb] = b1v[1 * H_ + c];
        bg1[hb] = b1v[2 * H_ + c]; bo1[hb] = b1v[3 * H_ + c];
        g0c[hb] = g0v[c]; be0c[hb] = be0v[c];
        g1c[hb] = g1v[c]; be1c[hb] = be1v[c];
    }

    float c0r[16], c1r[16];
#pragma unroll
    for (int k = 0; k < 16; ++k) { c0r[k] = 0.f; c1r[k] = 0.f; }

    __syncthreads();

    for (int t = 0; t < T_; ++t) {
        // stage x_t tile [16 x 64] -> f16 LDS ; prefetch next step's tile
        for (int i = tid; i < 16 * I_; i += 512) {
            int r = i >> 6, cc = i & 63;
            const float* src = &x[((size_t)(brow0 + r) * T_ + t) * I_ + cc];
            sxt[r * XTP + cc] = (_Float16)(*src);
            if (t + 1 < T_) __builtin_prefetch(src + I_, 0, 1);
        }
        __syncthreads();

        // ---- layer 0:  z0 = x_t @ Wx0 + h0 @ Wh0 + b0 ----
        v8f acc[8];
#pragma unroll
        for (int q = 0; q < 8; ++q) acc[q] = (v8f){0,0,0,0,0,0,0,0};
        mm_acc(acc, sxt, XTP, 2,  wx0p, w, lane);   // K = 64
        mm_acc(acc, sh0, LDP, 16, wh0p, w, lane);   // K = 512

        float h0n[16];
        gate_ln(acc, c0r, bi0, bf0, bg0, bo0, g0c, be0c,
                red_s0, red_q0, smean0, srstd0, w, lane, h0n);

        // publish h0 for layer-1 input and next-step recurrence
#pragma unroll
        for (int hb = 0; hb < 2; ++hb)
#pragma unroll
            for (int r = 0; r < 8; ++r) {
                int row = gg * 8 + r;
                sh0[row * LDP + col0 + hb * 16] = (_Float16)h0n[hb * 8 + r];
            }
        __syncthreads();

        // ---- layer 1:  z1 = h0n @ Wx1 + h1 @ Wh1 + b1 ----
#pragma unroll
        for (int q = 0; q < 8; ++q) acc[q] = (v8f){0,0,0,0,0,0,0,0};
        mm_acc(acc, sh0, LDP, 16, wx1p, w, lane);
        mm_acc(acc, sh1, LDP, 16, wh1p, w, lane);

        float h1n[16];
        gate_ln(acc, c1r, bi1, bf1, bg1, bo1, g1c, be1c,
                red_s1, red_q1, smean1, srstd1, w, lane, h1n);

#pragma unroll
        for (int k = 0; k < 16; ++k) h1n[k] += h0n[k];   // residual

        // publish h1 (LDS for recurrence, global f32 for attention)
#pragma unroll
        for (int hb = 0; hb < 2; ++hb)
#pragma unroll
            for (int r = 0; r < 8; ++r) {
                int row = gg * 8 + r;
                int c   = col0 + hb * 16;
                float v = h1n[hb * 8 + r];
                sh1[row * LDP + c] = (_Float16)v;
                hs[((size_t)(brow0 + row) * T_ + t) * H_ + c] = v;
            }
        __syncthreads();
    }
}

// ---------------------------------------------------------------------------
// Attention pooling + FC: one block per batch row.
// ---------------------------------------------------------------------------
__global__ __launch_bounds__(512)
void attention_kernel(const float* __restrict__ hs, const float* __restrict__ qv,
                      const float* __restrict__ fcw, const float* __restrict__ fcb,
                      float* __restrict__ out) {
    __shared__ float sc[T_];
    __shared__ float sq[H_];
    __shared__ float red[512];
    const int b = blockIdx.x, tid = threadIdx.x;
    const float* hb = hs + (size_t)b * T_ * H_;

    sq[tid] = qv[tid];
    __syncthreads();

    float s = 0.f;                               // score for time step `tid`
    for (int h = 0; h < H_; ++h) s += hb[(size_t)tid * H_ + h] * sq[h];

    red[tid] = s;
    __syncthreads();
    for (int off = 256; off > 0; off >>= 1) {
        if (tid < off) red[tid] = fmaxf(red[tid], red[tid + off]);
        __syncthreads();
    }
    float mx = red[0];
    __syncthreads();
    float e = __expf(s - mx);
    red[tid] = e;
    __syncthreads();
    for (int off = 256; off > 0; off >>= 1) {
        if (tid < off) red[tid] += red[tid + off];
        __syncthreads();
    }
    float inv = 1.f / red[0];
    __syncthreads();
    sc[tid] = e * inv;
    __syncthreads();

    float ctx = 0.f;                             // context component `tid`
    for (int t = 0; t < T_; ++t) ctx += sc[t] * hb[(size_t)t * H_ + tid];

    red[tid] = ctx * fcw[tid];
    __syncthreads();
    for (int off = 256; off > 0; off >>= 1) {
        if (tid < off) red[tid] += red[tid + off];
        __syncthreads();
    }
    if (tid == 0) out[b] = red[0] + fcb[0];
}

// ---------------------------------------------------------------------------
extern "C" void kernel_launch(void* const* d_in, const int* in_sizes, int n_in,
                              void* d_out, int out_size, void* d_ws, size_t ws_size,
                              hipStream_t stream) {
    const float* x   = (const float*)d_in[0];
    const float* Wx0 = (const float*)d_in[1];
    const float* Wh0 = (const float*)d_in[2];
    const float* b0  = (const float*)d_in[3];
    const float* g0  = (const float*)d_in[4];
    const float* be0 = (const float*)d_in[5];
    const float* Wx1 = (const float*)d_in[6];
    const float* Wh1 = (const float*)d_in[7];
    const float* b1  = (const float*)d_in[8];
    const float* g1  = (const float*)d_in[9];
    const float* be1 = (const float*)d_in[10];
    const float* q   = (const float*)d_in[11];
    const float* fcw = (const float*)d_in[12];
    const float* fcb = (const float*)d_in[13];
    float* out = (float*)d_out;

    char*  ws       = (char*)d_ws;
    size_t hs_bytes = (size_t)B_ * T_ * H_ * sizeof(float);   // 134 MB
    float*    hs   = (float*)ws;
    _Float16* wx0p = (_Float16*)(ws + hs_bytes);
    _Float16* wh0p = wx0p + (size_t)I_ * 4 * H_;
    _Float16* wx1p = wh0p + (size_t)H_ * 4 * H_;
    _Float16* wh1p = wx1p + (size_t)H_ * 4 * H_;

    const int NW = 4 * H_;   // 2048
    pack_weights_kernel<<<(I_ * NW + 255) / 256, 256, 0, stream>>>(Wx0, wx0p, I_, NW);
    pack_weights_kernel<<<(H_ * NW + 255) / 256, 256, 0, stream>>>(Wh0, wh0p, H_, NW);
    pack_weights_kernel<<<(H_ * NW + 255) / 256, 256, 0, stream>>>(Wx1, wx1p, H_, NW);
    pack_weights_kernel<<<(H_ * NW + 255) / 256, 256, 0, stream>>>(Wh1, wh1p, H_, NW);

    lstm_kernel<<<B_ / 16, 512, 0, stream>>>(x, wx0p, wh0p, wx1p, wh1p,
                                             b0, g0, be0, b1, g1, be1, hs);

    attention_kernel<<<B_, 512, 0, stream>>>(hs, q, fcw, fcb, out);
}
